// VerticalSelfAttentionBlock_39513699123688
// MI455X (gfx1250) — matrix operand order repaired and checked by reading, not compiled, not run on hardware
//
#include <hip/hip_runtime.h>

// ---------------------------------------------------------------------------
// VerticalSelfAttentionBlock for MI455X (gfx1250, wave32, WMMA bf16, TDM)
//
// Pipeline (all on `stream`):
//   1) ada = cond @ norm_w^T + 1                         (fp32, tiny)
//   2) convert qkv_w / out_w to bf16                     (bandwidth)
//   3) xn = rmsnorm(x) * ada  -> bf16                    (bandwidth)
//   4) qkv = xn @ qkv_w^T                                (WMMA bf16, TDM-fed)
//   5) fused cosine-sim attention per (n*w, head)        (WMMA bf16 + reg softmax)
//   6) out = o @ out_w^T + x                             (WMMA bf16, TDM-fed)
//
// Workspace requirement: ~680 MB.
// ---------------------------------------------------------------------------

typedef unsigned short u16;
typedef unsigned int   u32;
typedef unsigned long long u64;

typedef __attribute__((ext_vector_type(8)))  float  v8f;
typedef __attribute__((ext_vector_type(16))) __bf16 v16bf;

struct __attribute__((aligned(16))) U4 { u32 a, b, c, d; };
union FragBF { U4 u[2]; v16bf v; };

// Tensor-DMA path. This toolchain exposes the 6-arg (clang-23 / therock)
// builtin: (uint32x4 g0, int32x8 g1, int32x4 g2, int32x4 g3, int32x8 gx, i32 cpol).
// Fallback compiles the proven synchronous LDS-staging path.
#if defined(__HIP_DEVICE_COMPILE__) && defined(__gfx1250__) &&      \
    __has_builtin(__builtin_amdgcn_tensor_load_to_lds) &&           \
    __has_builtin(__builtin_amdgcn_s_wait_tensorcnt)
#define USE_TDM 1
#else
#define USE_TDM 0
#endif

// ----- sizes -----
#define NB   2
#define HH   128
#define WW   256
#define DD   1024
#define CF   768
#define NHD  16
#define DHD  64
#define MROW (NB * HH * WW)     // 65536
#define TQKV (3 * DD)           // 3072

// ----- workspace layout (bytes) -----
#define OFF_ADA   ((size_t)0)
#define OFF_WQKV  ((size_t)8192)
#define OFF_WOUT  ((size_t)8192 + 6291456)
#define OFF_XN    ((size_t)8396800)
#define OFF_QKV   ((size_t)142614528)
#define OFF_O     ((size_t)545267712)

// ----- scalar bf16 helpers -----
__device__ __forceinline__ float bf2f(u16 v) {
  return __uint_as_float((u32)v << 16);
}
__device__ __forceinline__ u16 f2bf(float f) {
  u32 u = __float_as_uint(f);
  return (u16)((u + 0x7FFFu + ((u >> 16) & 1u)) >> 16);
}
__device__ __forceinline__ v8f vzero8() {
  v8f z;
#pragma unroll
  for (int i = 0; i < 8; ++i) z[i] = 0.f;
  return z;
}
__device__ __forceinline__ v8f wmma_bf16(v16bf a, v16bf b, v8f c) {
  return __builtin_amdgcn_wmma_f32_16x16x32_bf16(
      false, a, false, b, (short)0, c, false, false);
}

// A-fragment (16x32 bf16): lane m = lane&15, half = lane>>4.
// VGPR0-3 hold k = 8*half..8*half+7 (packed pairs), VGPR4-7 hold k = 16+8*half..
__device__ __forceinline__ v16bf load_frag_a(const u16* lds, int stride,
                                             int row0, int kcol, int lane) {
  const int hl = lane >> 4, m = lane & 15;
  const u16* p = lds + (row0 + m) * stride + kcol + 8 * hl;
  FragBF f;
  f.u[0] = *(const U4*)p;
  f.u[1] = *(const U4*)(p + 16);
  return f.v;
}
// B-fragment (32x16 bf16): lane n = lane&15, half = lane>>4.
// VGPR0-3 hold k = 8*half..8*half+7 (packed pairs), VGPR4-7 hold k = 16+8*half..
__device__ __forceinline__ v16bf load_frag_b(const u16* lds, int stride,
                                             int col0, int kcol, int lane) {
  const int hl = lane >> 4, n = lane & 15;
  const u16* p = lds + (col0 + n) * stride + kcol + 8 * hl;
  FragBF f;
  f.u[0] = *(const U4*)p;
  f.u[1] = *(const U4*)(p + 16);
  return f.v;
}

#if USE_TDM
typedef __attribute__((ext_vector_type(4))) unsigned int v4u_t;
typedef __attribute__((ext_vector_type(8))) int          v8i_t;
typedef __attribute__((ext_vector_type(4))) int          v4i_t;
typedef __attribute__((address_space(3))) void           lds_void_t;

__device__ __forceinline__ u32 lds_off(void* p) {
  return (u32)(size_t)(lds_void_t*)p;
}

// Issue one TDM descriptor: load a 128(row) x 32(elem) bf16 tile from a
// row-major [*, row_stride] tensor into LDS, with HW padding of 4 DWORDs
// after every 16 DWORDs (64B row -> 80B LDS pitch = stride 40 u16).
// D# layout per CDNA5 ISA.
__device__ __forceinline__ void tdm_load_tile_128x32(const u16* gsrc,
                                                     u32 lds_byte,
                                                     u32 row_stride_elems) {
  const u64 ga = (u64)(size_t)gsrc;
  v4u_t g0;
  g0[0] = 1u;                                            // count=1 (valid D#)
  g0[1] = lds_byte;                                      // lds_addr
  g0[2] = (u32)ga;                                       // global_addr[31:0]
  g0[3] = (u32)((ga >> 32) & 0x01FFFFFFu) | (2u << 30);  // addr[56:32] | type=2

  const u32 w0 = (1u << 16)    // data_size = 2 bytes
               | (1u << 20)    // pad_enable
               | (3u << 22)    // pad_interval: 16 DWORDs
               | (3u << 25);   // pad_amount:   4 DWORDs
  const u32 tdim0 = row_stride_elems;   // tensor dim0 (elements)
  const u32 tdim1 = 0x40000000u;        // effectively unbounded (no OOB here)
  v8i_t g1;
  g1[0] = (int)w0;
  g1[1] = (int)((tdim0 & 0xFFFFu) << 16);                  // tdim0[15:0] @63:48
  g1[2] = (int)((tdim0 >> 16) | ((tdim1 & 0xFFFFu) << 16));// tdim0 hi | tdim1 lo
  g1[3] = (int)((tdim1 >> 16) | (32u << 16));              // tdim1 hi | tile_dim0=32
  g1[4] = (int)128u;                                       // tile_dim1=128, tile_dim2=0
  g1[5] = (int)row_stride_elems;                           // dim0_stride[31:0]
  g1[6] = 0;                                               // dim0_stride[47:32]
  g1[7] = 0;
  v4i_t gz4;
  gz4[0] = gz4[1] = gz4[2] = gz4[3] = 0;                   // 2-D tensor: groups 2/3 zero
  v8i_t gz8;
#pragma unroll
  for (int i = 0; i < 8; ++i) gz8[i] = 0;                  // extra clang-23 group: zero
  __builtin_amdgcn_tensor_load_to_lds(g0, g1, gz4, gz4, gz8, 0);
}
#endif  // USE_TDM

// ---------------------------------------------------------------------------
// 1) ada = cond @ norm_w^T + 1     [NB, DD]
// ---------------------------------------------------------------------------
__global__ __launch_bounds__(256) void ada_kernel(const float* __restrict__ cond,
                                                  const float* __restrict__ norm_w,
                                                  float* __restrict__ ada) {
  const int idx = blockIdx.x * 256 + threadIdx.x;   // NB*DD = 2048 threads
  if (idx >= NB * DD) return;
  const int n = idx >> 10, d = idx & (DD - 1);
  float s = 1.0f;
  for (int c = 0; c < CF; ++c)
    s = fmaf(cond[n * CF + c], norm_w[d * CF + c], s);
  ada[idx] = s;
}

// ---------------------------------------------------------------------------
// 2) fp32 -> bf16 convert
// ---------------------------------------------------------------------------
__global__ __launch_bounds__(256) void cvt_kernel(const float* __restrict__ in,
                                                  u16* __restrict__ out, int count) {
  const int idx = blockIdx.x * 256 + threadIdx.x;
  if (idx < count) out[idx] = f2bf(in[idx]);
}

// ---------------------------------------------------------------------------
// 3) xn = x * ada * rsqrt(mean(x^2) + eps)  -> bf16, one block per row
// ---------------------------------------------------------------------------
__global__ __launch_bounds__(256) void rmsnorm_kernel(const float* __restrict__ x,
                                                      const float* __restrict__ ada,
                                                      u16* __restrict__ xn) {
  __shared__ float red[256];
  const int row = blockIdx.x;             // 0..MROW-1
  const int n = row >> 15;                // row / (HH*WW)
  const size_t base = (size_t)row * DD;
  const int t = threadIdx.x;

  float vals[4];
  float ss = 0.f;
#pragma unroll
  for (int j = 0; j < 4; ++j) {
    vals[j] = x[base + t * 4 + j];
    ss = fmaf(vals[j], vals[j], ss);
  }
  red[t] = ss;
  __syncthreads();
  for (int s2 = 128; s2 > 0; s2 >>= 1) {
    if (t < s2) red[t] += red[t + s2];
    __syncthreads();
  }
  const float r = rsqrtf(red[0] * (1.f / (float)DD) + 1e-6f);
#pragma unroll
  for (int j = 0; j < 4; ++j) {
    const int col = t * 4 + j;
    xn[base + col] = f2bf(vals[j] * ada[(n << 10) + col] * r);
  }
}

// ---------------------------------------------------------------------------
// 4/6) Tiled WMMA GEMM:  C[M,N] = A[M,K] @ B[N,K]^T   (bf16 in, fp32 acc)
//      OUT_F32: fp32 output + residual; else bf16 output.
//      Block tile 128x128, 8 waves (4 along M x 2 along N), wave tile 32x64.
//      TDM path: BK=32, double-buffered, wave0 drives the Tensor Data Mover.
// ---------------------------------------------------------------------------
template <bool OUT_F32>
__global__ __launch_bounds__(256) void gemm_bf16_kernel(const u16* __restrict__ A,
                                                        const u16* __restrict__ B,
                                                        void* __restrict__ Cv,
                                                        const float* __restrict__ resid,
                                                        int M, int N, int K) {
  const int tid = threadIdx.x;
  const int lane = tid & 31, wid = tid >> 5;
  const int wm = wid & 3, wn = wid >> 2;
  const int m0 = blockIdx.y * 128, n0 = blockIdx.x * 128;

  v8f acc[2][4];
#pragma unroll
  for (int i = 0; i < 2; ++i)
#pragma unroll
    for (int j = 0; j < 4; ++j) acc[i][j] = vzero8();

#if USE_TDM
  // ---- TDM-fed, double-buffered (BK=32, LDS pitch 40 u16 via HW padding) ----
  __shared__ u16 lA[2][128 * 40];
  __shared__ u16 lB[2][128 * 40];
  const u32 ldsA[2] = {lds_off(&lA[0][0]), lds_off(&lA[1][0])};
  const u32 ldsB[2] = {lds_off(&lB[0][0]), lds_off(&lB[1][0])};

  if (tid < 32) {   // wave 0 drives the DMA; EXEC-independent issue, 1 wave only
    tdm_load_tile_128x32(A + (size_t)m0 * K, ldsA[0], (u32)K);
    tdm_load_tile_128x32(B + (size_t)n0 * K, ldsB[0], (u32)K);
    __builtin_amdgcn_s_wait_tensorcnt(0);
  }
  __syncthreads();

  int buf = 0;
  for (int kk = 0; kk < K; kk += 32) {
    const bool more = (kk + 32) < K;
    if (more && tid < 32) {   // prefetch next stage while this one computes
      tdm_load_tile_128x32(A + (size_t)m0 * K + kk + 32, ldsA[buf ^ 1], (u32)K);
      tdm_load_tile_128x32(B + (size_t)n0 * K + kk + 32, ldsB[buf ^ 1], (u32)K);
    }
    const u16* cA = &lA[buf][0];
    const u16* cB = &lB[buf][0];
    v16bf a[2], b[4];
#pragma unroll
    for (int i = 0; i < 2; ++i)
      a[i] = load_frag_a(cA, 40, wm * 32 + i * 16, 0, lane);
#pragma unroll
    for (int j = 0; j < 4; ++j)
      b[j] = load_frag_b(cB, 40, wn * 64 + j * 16, 0, lane);
#pragma unroll
    for (int i = 0; i < 2; ++i)
#pragma unroll
      for (int j = 0; j < 4; ++j)
        acc[i][j] = wmma_bf16(a[i], b[j], acc[i][j]);
    if (more && tid < 32) __builtin_amdgcn_s_wait_tensorcnt(0);
    __syncthreads();
    buf ^= 1;
  }
#else
  // ---- synchronous staging fallback (BK=64, pitch 72 u16) ----
  __shared__ u16 lA[128 * 72];
  __shared__ u16 lB[128 * 72];
  const int lr = tid >> 3;           // 0..31
  const int lc = (tid & 7) * 8;      // ushort column (16B chunks)

  for (int kk = 0; kk < K; kk += 64) {
#pragma unroll
    for (int p = 0; p < 4; ++p) {
      const int r = lr + p * 32;
      *(U4*)(lA + r * 72 + lc) = *(const U4*)(A + (size_t)(m0 + r) * K + kk + lc);
      *(U4*)(lB + r * 72 + lc) = *(const U4*)(B + (size_t)(n0 + r) * K + kk + lc);
    }
    if (kk + 64 < K) {   // speculative L2 prefetch of next K slab
      __builtin_prefetch(A + (size_t)(m0 + lr) * K + kk + 64 + lc, 0, 1);
      __builtin_prefetch(B + (size_t)(n0 + lr) * K + kk + 64 + lc, 0, 1);
    }
    __syncthreads();
#pragma unroll
    for (int ks = 0; ks < 2; ++ks) {
      v16bf a[2], b[4];
#pragma unroll
      for (int i = 0; i < 2; ++i)
        a[i] = load_frag_a(lA, 72, wm * 32 + i * 16, ks * 32, lane);
#pragma unroll
      for (int j = 0; j < 4; ++j)
        b[j] = load_frag_b(lB, 72, wn * 64 + j * 16, ks * 32, lane);
#pragma unroll
      for (int i = 0; i < 2; ++i)
#pragma unroll
        for (int j = 0; j < 4; ++j)
          acc[i][j] = wmma_bf16(a[i], b[j], acc[i][j]);
    }
    __syncthreads();
  }
#endif

  // epilogue: C layout -> m = 8*half + vgpr, n = lane&15
  const int hl = lane >> 4, nl = lane & 15;
#pragma unroll
  for (int i = 0; i < 2; ++i)
#pragma unroll
    for (int j = 0; j < 4; ++j) {
      const int gmb = m0 + wm * 32 + i * 16 + 8 * hl;
      const int gn = n0 + wn * 64 + j * 16 + nl;
#pragma unroll
      for (int e = 0; e < 8; ++e) {
        const size_t idx = (size_t)(gmb + e) * N + gn;
        const float v = acc[i][j][e];
        if (OUT_F32)
          ((float*)Cv)[idx] = v + resid[idx];
        else
          ((u16*)Cv)[idx] = f2bf(v);
      }
    }
}

// ---------------------------------------------------------------------------
// 5) Fused attention per (b = n*WW + w, head). 256 threads = 8 waves.
//    seq = HH = 128 (the 'h' axis), dh = 64.
//    LDS: Q[128x64] (stride 72) | K[128x64] (stride 72) | V^T[64x128] (stride 136)
//    P[128x128] (stride 136) overlays Q/K after a barrier.
// ---------------------------------------------------------------------------
__global__ __launch_bounds__(256) void attn_kernel(const u16* __restrict__ qkv,
                                                   const float* __restrict__ scale,
                                                   u16* __restrict__ o) {
  __shared__ u16 smem[27136];          // 54272 B
  u16* lQ = smem;                      // 128 * 72
  u16* lK = smem + 9216;               // 128 * 72
  u16* lV = smem + 18432;              // 64  * 136  (V transposed: [e][j])
  u16* lP = smem;                      // 128 * 136  (overlay Q+K)

  const int head  = blockIdx.x & (NHD - 1);
  const int b     = blockIdx.x >> 4;
  const int n_idx = b >> 8;            // b / WW
  const int w_idx = b & (WW - 1);

  const int tid  = threadIdx.x;
  const int lane = tid & 31, wid = tid >> 5;

  // ---- load + cosine-normalize Q,K; load V transposed ----
  {
    const int row  = tid >> 1;         // h position, 0..127
    const int part = tid & 1;          // which 32-wide half of dh
    const size_t rb =
        ((size_t)(n_idx * HH + row) * WW + w_idx) * TQKV + head * DHD + part * 32;
    const float sscale = sqrtf(scale[head]);

    float qv[32];
    float ssq = 0.f;
#pragma unroll
    for (int j = 0; j < 32; ++j) {
      const float f = bf2f(qkv[rb + j]);
      qv[j] = f;
      ssq = fmaf(f, f, ssq);
    }
    ssq += __shfl_xor(ssq, 1);
    float rn = sscale * rsqrtf(ssq + 1e-6f);
#pragma unroll
    for (int j = 0; j < 32; ++j) lQ[row * 72 + part * 32 + j] = f2bf(qv[j] * rn);

    ssq = 0.f;
#pragma unroll
    for (int j = 0; j < 32; ++j) {
      const float f = bf2f(qkv[rb + DD + j]);
      qv[j] = f;
      ssq = fmaf(f, f, ssq);
    }
    ssq += __shfl_xor(ssq, 1);
    rn = sscale * rsqrtf(ssq + 1e-6f);
#pragma unroll
    for (int j = 0; j < 32; ++j) lK[row * 72 + part * 32 + j] = f2bf(qv[j] * rn);

#pragma unroll
    for (int j = 0; j < 32; ++j)
      lV[(part * 32 + j) * 136 + row] = qkv[rb + 2 * DD + j];
  }
  __syncthreads();

  // ---- S = Q @ K^T : wave owns query rows m0..m0+15, all 128 key columns ----
  const int m0 = wid * 16;
  v8f s[8];
#pragma unroll
  for (int j = 0; j < 8; ++j) s[j] = vzero8();
#pragma unroll
  for (int ks = 0; ks < 2; ++ks) {
    const v16bf a = load_frag_a(lQ, 72, m0, ks * 32, lane);
#pragma unroll
    for (int j = 0; j < 8; ++j) {
      const v16bf bf = load_frag_b(lK, 72, j * 16, ks * 32, lane);
      s[j] = wmma_bf16(a, bf, s[j]);
    }
  }

  // ---- softmax per row (row spread over 16 lanes in C layout) ----
  float inv_row[8];
#pragma unroll
  for (int i = 0; i < 8; ++i) {
    float mx = s[0][i];
#pragma unroll
    for (int j = 1; j < 8; ++j) mx = fmaxf(mx, s[j][i]);
    for (int m = 1; m < 16; m <<= 1) mx = fmaxf(mx, __shfl_xor(mx, m));
    float sum = 0.f;
#pragma unroll
    for (int j = 0; j < 8; ++j) {
      const float p = __expf(s[j][i] - mx);
      s[j][i] = p;
      sum += p;
    }
    for (int m = 1; m < 16; m <<= 1) sum += __shfl_xor(sum, m);
    inv_row[i] = 1.f / sum;
  }

  __syncthreads();   // all waves done reading lQ/lK -> safe to overlay with P

  const int hl = lane >> 4, nl = lane & 15;
#pragma unroll
  for (int i = 0; i < 8; ++i) {
    const int r = m0 + 8 * hl + i;
#pragma unroll
    for (int j = 0; j < 8; ++j)
      lP[r * 136 + j * 16 + nl] = f2bf(s[j][i] * inv_row[i]);
  }
  // Each wave reads only its own P rows -> in-wave LDS ordering suffices.

  // ---- O = P @ V : 16 rows x 64 cols per wave, K = 128 ----
  v8f ov[4];
#pragma unroll
  for (int j = 0; j < 4; ++j) ov[j] = vzero8();
#pragma unroll
  for (int ks = 0; ks < 4; ++ks) {
    const v16bf a = load_frag_a(lP, 136, m0, ks * 32, lane);
#pragma unroll
    for (int j = 0; j < 4; ++j) {
      const v16bf bf = load_frag_b(lV, 136, j * 16, ks * 32, lane);
      ov[j] = wmma_bf16(a, bf, ov[j]);
    }
  }

  // ---- store O -> o[(n,h,w), head*64 + e] (bf16) ----
#pragma unroll
  for (int j = 0; j < 4; ++j) {
    const int gcol = head * DHD + j * 16 + nl;
#pragma unroll
    for (int i = 0; i < 8; ++i) {
      const int h = m0 + 8 * hl + i;
      const size_t row = (size_t)(n_idx * HH + h) * WW + w_idx;
      o[row * DD + gcol] = f2bf(ov[j][i]);
    }
  }
}

// ---------------------------------------------------------------------------
// launcher
// ---------------------------------------------------------------------------
extern "C" void kernel_launch(void* const* d_in, const int* in_sizes, int n_in,
                              void* d_out, int out_size, void* d_ws, size_t ws_size,
                              hipStream_t stream) {
  const float* x      = (const float*)d_in[0];
  const float* cond   = (const float*)d_in[1];
  const float* norm_w = (const float*)d_in[2];
  const float* qkv_w  = (const float*)d_in[3];
  const float* scale  = (const float*)d_in[4];
  const float* out_w  = (const float*)d_in[5];
  float* out = (float*)d_out;

  char* ws = (char*)d_ws;
  float* ada    = (float*)(ws + OFF_ADA);
  u16*   wqkv   = (u16*)(ws + OFF_WQKV);
  u16*   wout   = (u16*)(ws + OFF_WOUT);
  u16*   xn     = (u16*)(ws + OFF_XN);
  u16*   qkv_ws = (u16*)(ws + OFF_QKV);
  u16*   o_ws   = (u16*)(ws + OFF_O);

  // weight conversions
  cvt_kernel<<<(TQKV * DD + 255) / 256, 256, 0, stream>>>(qkv_w, wqkv, TQKV * DD);
  cvt_kernel<<<(DD * DD + 255) / 256, 256, 0, stream>>>(out_w, wout, DD * DD);

  // ada
  ada_kernel<<<(NB * DD + 255) / 256, 256, 0, stream>>>(cond, norm_w, ada);

  // rmsnorm + bf16 cast
  rmsnorm_kernel<<<MROW, 256, 0, stream>>>(x, ada, xn);

  // qkv = xn @ qkv_w^T   [65536 x 3072]
  gemm_bf16_kernel<false><<<dim3(TQKV / 128, MROW / 128), 256, 0, stream>>>(
      xn, wqkv, (void*)qkv_ws, nullptr, MROW, TQKV, DD);

  // fused attention (512 batches x 16 heads)
  attn_kernel<<<NB * WW * NHD, 256, 0, stream>>>(qkv_ws, scale, o_ws);

  // out = o @ out_w^T + x   [65536 x 1024] fp32
  gemm_bf16_kernel<true><<<dim3(DD / 128, MROW / 128), 256, 0, stream>>>(
      o_ws, wout, (void*)out, x, MROW, DD, DD);
}
